// SparseAttention_32263794327902
// MI455X (gfx1250) — matrix operand order repaired and checked by reading, not compiled, not run on hardware
//
#include <hip/hip_runtime.h>

// ---------------------------------------------------------------------------
// MI455X (gfx1250) local-window causal attention, bf16 WMMA implementation.
// B=2, S=2048, D=1024, H=16, HEAD_DIM=64, HALF_WINDOW=32, SCALE=1/8.
//
// Roofline: 4 GEMMs of 4096x1024x1024 (~34 GFLOP) dominate; all data (~50MB)
// is L2-resident (192MB), HBM time ~2us @23.3TB/s -> matrix-core bound.
// Hot loops run on v_wmma_f32_16x16x32_bf16 (f32 accumulate).  LDS staging
// uses the CDNA5 async path (global_load_async_to_lds_*, ASYNCcnt) with a
// double-buffered ping-pong so the matrix pipe overlaps the tile stream.
// Attention exploits the 33-wide band: 48 keys per 16-query tile instead of
// the reference's dense 2048x2048 score matrix.
// ---------------------------------------------------------------------------

typedef __attribute__((ext_vector_type(16))) __bf16 bf16x16;
typedef __attribute__((ext_vector_type(8)))  __bf16 bf16x8;
typedef __attribute__((ext_vector_type(4)))  __bf16 bf16x4;
typedef __attribute__((ext_vector_type(8)))  float  f32x8;

#define S_LEN 2048
#define D_DIM 1024
#define NHEAD 16
#define HDIM  64
#define MROWS 4096   // B*S

// ---- CDNA5 async LDS staging (ASYNCcnt) -----------------------------------
__device__ __forceinline__ void async_lds_b64(const void* gsrc, void* lds_dst) {
  unsigned lds = (unsigned)(uintptr_t)lds_dst;          // low 32 = LDS offset
  unsigned long long ga = (unsigned long long)(uintptr_t)gsrc;
  asm volatile("global_load_async_to_lds_b64 %0, %1, off"
               :: "v"(lds), "v"(ga) : "memory");
}
__device__ __forceinline__ void async_lds_b128(const void* gsrc, void* lds_dst) {
  unsigned lds = (unsigned)(uintptr_t)lds_dst;
  unsigned long long ga = (unsigned long long)(uintptr_t)gsrc;
  asm volatile("global_load_async_to_lds_b128 %0, %1, off"
               :: "v"(lds), "v"(ga) : "memory");
}
__device__ __forceinline__ void wait_async0() {
  asm volatile("s_wait_asynccnt 0" ::: "memory");
}

// ---- WMMA fragment helpers (contiguous k order: k = 16*(lane>>4) + s) ------
__device__ __forceinline__ bf16x16 frag_join(bf16x8 lo, bf16x8 hi) {
  bf16x16 r;
#pragma unroll
  for (int i = 0; i < 8; ++i) { r[i] = lo[i]; r[i + 8] = hi[i]; }
  return r;
}

// A-fragment (16x32 bf16, MxK), tile row-major [16][stride].
__device__ __forceinline__ bf16x16 load_a_frag(const __bf16* t, int stride,
                                               int k0, int lane) {
  const int m = lane & 15, h = lane >> 4;
  const __bf16* p = t + m * stride + k0 + h * 16;
  return frag_join(*(const bf16x8*)p, *(const bf16x8*)(p + 8));
}

// B-fragment (32x16 bf16, KxN) from "n-major" storage (rows are output cols):
// lane n holds B[k][n] for k = k0 + (lane>>4)*16 .. +15  (16 contiguous).
__device__ __forceinline__ bf16x16 load_b_frag(const __bf16* t, int stride,
                                               int nbase, int k0, int lane) {
  const int n = lane & 15, h = lane >> 4;
  const __bf16* p = t + (size_t)(nbase + n) * stride + k0 + h * 16;
  return frag_join(*(const bf16x8*)p, *(const bf16x8*)(p + 8));
}

// B-fragment from K-major (row-major [k][n]) storage: strided column reads.
__device__ __forceinline__ bf16x16 load_b_frag_cm(const __bf16* t, int stride,
                                                  int nbase, int k0, int lane) {
  const int n = lane & 15, h = lane >> 4;
  const __bf16* p = t + (size_t)(k0 + h * 16) * stride + nbase + n;
  bf16x16 r;
#pragma unroll
  for (int i = 0; i < 16; ++i) r[i] = p[(size_t)i * stride];
  return r;
}

__device__ __forceinline__ f32x8 wmma_bf16(bf16x16 a, bf16x16 b, f32x8 c) {
  return __builtin_amdgcn_wmma_f32_16x16x32_bf16(
      false, a, false, b, (short)0, c, false, false);
}

// ---------------------------------------------------------------------------
// Stage 0: precision casts
// ---------------------------------------------------------------------------
__global__ void cast_bf16_kernel(const float* __restrict__ in,
                                 __bf16* __restrict__ out, int n) {
  int i = blockIdx.x * blockDim.x + threadIdx.x;
  if (i < n) out[i] = (__bf16)in[i];
}

// W [K=1024][N=1024] row-major  ->  WT [N][K] bf16 (for contiguous B-frags).
__global__ void transpose_cast_kernel(const float* __restrict__ in,
                                      __bf16* __restrict__ out) {
  int i = blockIdx.x * blockDim.x + threadIdx.x;   // 1M threads
  int k = i >> 10, n = i & 1023;
  out[(size_t)n * D_DIM + k] = (__bf16)in[i];
}

// ---------------------------------------------------------------------------
// Stage 1: QKV projection GEMM.  Block = 4 waves, tile 16 rows x 256 cols.
// A tile double-buffered in LDS via async loads; B frags stream from L2.
// Out goes to head-major [B,H,S,64] bf16; Q is pre-scaled by 1/8.
// ---------------------------------------------------------------------------
__global__ __launch_bounds__(128)
void gemm_qkv_kernel(const __bf16* __restrict__ X,    // [4096][1024] bf16
                     const __bf16* __restrict__ WT,   // [1024 n][1024 k] bf16
                     __bf16* __restrict__ Out,        // [B,H,S,64] bf16
                     float scale) {
  __shared__ __bf16 at[2][16][48];                    // 16x32 A tiles (+pad)
  const int lane  = threadIdx.x & 31;
  const int wave  = threadIdx.x >> 5;
  const int rowb  = blockIdx.x;                       // 16-row block
  const int nbase = blockIdx.y * 256 + wave * 64;

  // per-thread A staging slot: 4 elems (8B) per K-step
  const int aidx = threadIdx.x * 4;
  const int ar = aidx >> 5, ac = aidx & 31;
  const __bf16* arow = X + (size_t)(rowb * 16 + ar) * D_DIM + ac;

  async_lds_b64(arow, &at[0][ar][ac]);                // prologue: buffer 0

  f32x8 acc[4] = {};
  for (int k0 = 0; k0 < D_DIM; k0 += 32) {
    const int cur = (k0 >> 5) & 1;
    wait_async0();                                    // my async writes done
    __syncthreads();                                  // everyone's done
    if (k0 + 32 < D_DIM)                              // prefetch next tile
      async_lds_b64(arow + k0 + 32, &at[cur ^ 1][ar][ac]);
    if (k0 + 64 < D_DIM)                              // L2 prefetch of weights
      __builtin_prefetch(WT + (size_t)(nbase + lane) * D_DIM + k0 + 64, 0, 1);

    bf16x16 a = load_a_frag(&at[cur][0][0], 48, 0, lane);
#pragma unroll
    for (int t = 0; t < 4; ++t) {
      bf16x16 b = load_b_frag(WT, D_DIM, nbase + t * 16, k0, lane);
      acc[t] = wmma_bf16(a, b, acc[t]);
    }
  }

  // epilogue: permute into [B,H,S,64] while storing
  const int n0 = lane & 15, hf = lane >> 4;
#pragma unroll
  for (int t = 0; t < 4; ++t)
#pragma unroll
    for (int r = 0; r < 8; ++r) {
      int row = rowb * 16 + hf * 8 + r;               // b*S + s
      int col = nbase + t * 16 + n0;                  // h*64 + d
      int b = row >> 11, s = row & (S_LEN - 1);
      int h = col >> 6,  d = col & 63;
      Out[(((size_t)(b * NHEAD + h) * S_LEN) + s) * HDIM + d] =
          (__bf16)(acc[t][r] * scale);
    }
}

// ---------------------------------------------------------------------------
// Stage 2: banded causal attention. One wave per (b,h,16-query tile).
// Window: keys j in [q0-32, q0+15] -> 48 keys -> 3 key tiles.
// Q/K/V staged to LDS via async B128 loads.
// ---------------------------------------------------------------------------
__global__ __launch_bounds__(32)
void attn_local_kernel(const __bf16* __restrict__ Q,
                       const __bf16* __restrict__ K,
                       const __bf16* __restrict__ V,
                       __bf16* __restrict__ O) {     // [B,S,1024] bf16
  __shared__ __bf16 qt[16][80];     // Q tile   [q][d]   (pad -> 16B aligned)
  __shared__ __bf16 kt[48][80];     // K tile   [key][d]
  __shared__ __bf16 vrow[64][72];   // V tile   [key][d], keys padded to 64
  __shared__ __bf16 pt[16][80];     // probabilities [q][key], cols 48..63 = 0

  const int lane = threadIdx.x;
  const int blk  = blockIdx.x;            // (b*H + h)*128 + qblock
  const int qb   = blk & 127;
  const int bh   = blk >> 7;              // b*H + h
  const int q0   = qb * 16;
  const int kstart = q0 - 32;

  const __bf16* Qh = Q + (size_t)bh * S_LEN * HDIM;
  const __bf16* Kh = K + (size_t)bh * S_LEN * HDIM;
  const __bf16* Vh = V + (size_t)bh * S_LEN * HDIM;

  // stage Q (16x64): async B128, 4 iters x 32 lanes x 8 elems
#pragma unroll
  for (int i = 0; i < 4; ++i) {
    int idx = (i * 32 + lane) * 8;
    int r = idx >> 6, c = idx & 63;
    async_lds_b128(Qh + (size_t)(q0 + r) * HDIM + c, &qt[r][c]);
  }
  // stage K (48x64). Rows with j<0 are left as garbage: their score columns
  // are replaced wholesale by the mask select, so NaN garbage is harmless.
#pragma unroll
  for (int i = 0; i < 12; ++i) {
    int idx = (i * 32 + lane) * 8;
    int r = idx >> 6, c = idx & 63;
    int j = kstart + r;
    if (j >= 0) async_lds_b128(Kh + (size_t)j * HDIM + c, &kt[r][c]);
  }
  // stage V (64 keys x 64 dims), row-major. OOB rows MUST be zero: a NaN
  // garbage value times a zero probability would still poison the WMMA dot.
#pragma unroll
  for (int i = 0; i < 16; ++i) {
    int idx = (i * 32 + lane) * 8;
    int r = idx >> 6, c = idx & 63;
    int j = kstart + r;
    if (j >= 0 && j < S_LEN)
      async_lds_b128(Vh + (size_t)j * HDIM + c, &vrow[r][c]);
    else
      *(bf16x8*)&vrow[r][c] = (bf16x8){};
  }
  // zero the padded probability columns 48..63
#pragma unroll
  for (int i = 0; i < 8; ++i) {
    int idx = i * 32 + lane;
    pt[idx >> 4][48 + (idx & 15)] = (__bf16)0.0f;
  }
  wait_async0();
  __syncthreads();

  // scores: S(16x48) = Q(16x64) x K^T, 2 WMMAs per key tile (K=32 each)
  bf16x16 a0 = load_a_frag(&qt[0][0], 80, 0, lane);
  bf16x16 a1 = load_a_frag(&qt[0][0], 80, 32, lane);
  f32x8 sc[3];
#pragma unroll
  for (int t = 0; t < 3; ++t) {
    f32x8 c = {};
    c = wmma_bf16(a0, load_b_frag(&kt[0][0], 80, t * 16, 0, lane), c);
    c = wmma_bf16(a1, load_b_frag(&kt[0][0], 80, t * 16, 32, lane), c);
    sc[t] = c;
  }

  // masked softmax over 48 keys.  C layout: lane holds col n for 8 rows;
  // a row's 16 cols per tile live in lanes differing in bits 0..3 only.
  const int n = lane & 15, hf = lane >> 4;
#pragma unroll
  for (int r = 0; r < 8; ++r) {
    const int qi = q0 + hf * 8 + r;
    float v[3], e[3], mx = -1e30f;
#pragma unroll
    for (int t = 0; t < 3; ++t) {
      int j = kstart + t * 16 + n;
      bool ok = (j >= 0) && (j <= qi) && ((qi - j) <= 32);
      v[t] = ok ? sc[t][r] : -1e30f;
      mx = fmaxf(mx, v[t]);
    }
#pragma unroll
    for (int m = 1; m < 16; m <<= 1) mx = fmaxf(mx, __shfl_xor(mx, m, 32));
    float sum = 0.0f;
#pragma unroll
    for (int t = 0; t < 3; ++t) { e[t] = __expf(v[t] - mx); sum += e[t]; }
#pragma unroll
    for (int m = 1; m < 16; m <<= 1) sum += __shfl_xor(sum, m, 32);
    const float inv = 1.0f / sum;
#pragma unroll
    for (int t = 0; t < 3; ++t)
      pt[hf * 8 + r][t * 16 + n] = (__bf16)(e[t] * inv);
  }
  __syncthreads();

  // O(16x64) = P(16x64, zero-padded) x V(64x64): 2 K-chunks x 4 N tiles.
  // V is key-major in LDS -> column-major B-frag reads.
  bf16x16 pa0 = load_a_frag(&pt[0][0], 80, 0, lane);
  bf16x16 pa1 = load_a_frag(&pt[0][0], 80, 32, lane);
  const int bq = bh >> 4, hh = bh & 15;
#pragma unroll
  for (int t = 0; t < 4; ++t) {
    f32x8 c = {};
    c = wmma_bf16(pa0, load_b_frag_cm(&vrow[0][0], 72, t * 16, 0, lane), c);
    c = wmma_bf16(pa1, load_b_frag_cm(&vrow[0][0], 72, t * 16, 32, lane), c);
#pragma unroll
    for (int r = 0; r < 8; ++r) {
      int s = q0 + hf * 8 + r;
      O[((size_t)bq * S_LEN + s) * D_DIM + hh * HDIM + t * 16 + n] =
          (__bf16)c[r];
    }
  }
}

// ---------------------------------------------------------------------------
// Stage 3: output projection GEMM + bias, f32 store.
// ---------------------------------------------------------------------------
__global__ __launch_bounds__(128)
void gemm_out_kernel(const __bf16* __restrict__ X,   // [4096][1024] bf16
                     const __bf16* __restrict__ WT,  // WoT [n][k] bf16
                     const float* __restrict__ bias,
                     float* __restrict__ Out) {      // [4096][1024] f32
  __shared__ __bf16 at[2][16][48];
  const int lane  = threadIdx.x & 31;
  const int wave  = threadIdx.x >> 5;
  const int rowb  = blockIdx.x;
  const int nbase = blockIdx.y * 256 + wave * 64;

  const int aidx = threadIdx.x * 4;
  const int ar = aidx >> 5, ac = aidx & 31;
  const __bf16* arow = X + (size_t)(rowb * 16 + ar) * D_DIM + ac;

  async_lds_b64(arow, &at[0][ar][ac]);

  f32x8 acc[4] = {};
  for (int k0 = 0; k0 < D_DIM; k0 += 32) {
    const int cur = (k0 >> 5) & 1;
    wait_async0();
    __syncthreads();
    if (k0 + 32 < D_DIM)
      async_lds_b64(arow + k0 + 32, &at[cur ^ 1][ar][ac]);
    if (k0 + 64 < D_DIM)
      __builtin_prefetch(WT + (size_t)(nbase + lane) * D_DIM + k0 + 64, 0, 1);

    bf16x16 a = load_a_frag(&at[cur][0][0], 48, 0, lane);
#pragma unroll
    for (int t = 0; t < 4; ++t) {
      bf16x16 b = load_b_frag(WT, D_DIM, nbase + t * 16, k0, lane);
      acc[t] = wmma_bf16(a, b, acc[t]);
    }
  }

  const int n0 = lane & 15, hf = lane >> 4;
#pragma unroll
  for (int t = 0; t < 4; ++t)
#pragma unroll
    for (int r = 0; r < 8; ++r) {
      int row = rowb * 16 + hf * 8 + r;
      int col = nbase + t * 16 + n0;
      Out[(size_t)row * D_DIM + col] = acc[t][r] + bias[col];
    }
}

// ---------------------------------------------------------------------------
extern "C" void kernel_launch(void* const* d_in, const int* in_sizes, int n_in,
                              void* d_out, int out_size, void* d_ws,
                              size_t ws_size, hipStream_t stream) {
  (void)in_sizes; (void)n_in; (void)out_size; (void)ws_size;
  const float* x  = (const float*)d_in[0];
  const float* Wq = (const float*)d_in[1];
  const float* Wk = (const float*)d_in[2];
  const float* Wv = (const float*)d_in[3];
  const float* Wo = (const float*)d_in[4];
  const float* bo = (const float*)d_in[5];
  float* out = (float*)d_out;

  char* ws = (char*)d_ws;
  __bf16* xb  = (__bf16*)ws; ws += (size_t)MROWS * D_DIM * 2;
  __bf16* wqT = (__bf16*)ws; ws += (size_t)D_DIM * D_DIM * 2;
  __bf16* wkT = (__bf16*)ws; ws += (size_t)D_DIM * D_DIM * 2;
  __bf16* wvT = (__bf16*)ws; ws += (size_t)D_DIM * D_DIM * 2;
  __bf16* woT = (__bf16*)ws; ws += (size_t)D_DIM * D_DIM * 2;
  __bf16* Qh  = (__bf16*)ws; ws += (size_t)MROWS * D_DIM * 2;
  __bf16* Kh  = (__bf16*)ws; ws += (size_t)MROWS * D_DIM * 2;
  __bf16* Vh  = (__bf16*)ws; ws += (size_t)MROWS * D_DIM * 2;
  __bf16* Ah  = (__bf16*)ws; ws += (size_t)MROWS * D_DIM * 2;

  const int nx = MROWS * D_DIM;
  cast_bf16_kernel<<<(nx + 255) / 256, 256, 0, stream>>>(x, xb, nx);
  const int nw = D_DIM * D_DIM;
  transpose_cast_kernel<<<nw / 256, 256, 0, stream>>>(Wq, wqT);
  transpose_cast_kernel<<<nw / 256, 256, 0, stream>>>(Wk, wkT);
  transpose_cast_kernel<<<nw / 256, 256, 0, stream>>>(Wv, wvT);
  transpose_cast_kernel<<<nw / 256, 256, 0, stream>>>(Wo, woT);

  dim3 g(MROWS / 16, D_DIM / 256), blk(128);
  gemm_qkv_kernel<<<g, blk, 0, stream>>>(xb, wqT, Qh, 0.125f);  // SCALE=1/8
  gemm_qkv_kernel<<<g, blk, 0, stream>>>(xb, wkT, Kh, 1.0f);
  gemm_qkv_kernel<<<g, blk, 0, stream>>>(xb, wvT, Vh, 1.0f);

  attn_local_kernel<<<2 * NHEAD * (S_LEN / 16), 32, 0, stream>>>(Qh, Kh, Vh, Ah);

  gemm_out_kernel<<<g, blk, 0, stream>>>(Ah, woT, bo, out);
}
